// SwinTransformerBlock_5669356836086
// MI455X (gfx1250) — compile-verified, matrix operand order checked by reading, not run on hardware
//
#include <hip/hip_runtime.h>

typedef __attribute__((ext_vector_type(16))) __bf16 v16bf;
typedef __attribute__((ext_vector_type(8)))  float  v8f;
typedef unsigned short u16t;

#define DIMC   256
#define NHEAD  8
#define HDIM   32
#define NWIN   49
#define ROWS   100352   // B * NW * N = 32*64*49
#define HID    1024
#define QSCALE 0.17677669529663687f
#define GCONST 0.7978845608028654f
#define LNEPS  1e-5f

__device__ __forceinline__ u16t f2bf(float f) {
    unsigned int u = __float_as_uint(f);
    u += 0x7FFFu + ((u >> 16) & 1u);      // round-to-nearest-even
    return (u16t)(u >> 16);
}

union Frag {
    v16bf v;
    uint4 q[2];
};

__device__ __forceinline__ Frag load_frag(const u16t* p) {
    Frag f;
    f.q[0] = *(const uint4*)(p);
    f.q[1] = *(const uint4*)(p + 16);
    return f;
}
__device__ __forceinline__ Frag zero_frag() {
    Frag f;
    f.q[0] = make_uint4(0u, 0u, 0u, 0u);
    f.q[1] = make_uint4(0u, 0u, 0u, 0u);
    return f;
}
__device__ __forceinline__ v8f zero_acc() {
    v8f z;
#pragma unroll
    for (int i = 0; i < 8; ++i) z[i] = 0.f;
    return z;
}
__device__ __forceinline__ float gelu_f(float x) {
    return 0.5f * x * (1.f + tanhf(GCONST * (x + 0.044715f * x * x * x)));
}

// -------------------------------------------------------------------------
// Weight convert + transpose: dst[n*K+k] = bf16(src[k*N+n])
// -------------------------------------------------------------------------
__global__ __launch_bounds__(256) void transpose_w(const float* __restrict__ src,
                                                   u16t* __restrict__ dst,
                                                   int K, int N) {
    int idx = blockIdx.x * 256 + threadIdx.x;
    if (idx < K * N) {
        int n = idx / K;
        int k = idx - n * K;
        dst[idx] = f2bf(src[(size_t)k * N + n]);
    }
}

// -------------------------------------------------------------------------
// LayerNorm (block per row, 256 threads = 1 elem/thread).
// WINDOWED=1: fuses cyclic shift (-3,-3) + 7x7 window partition for LN1.
// -------------------------------------------------------------------------
template <int WINDOWED>
__global__ __launch_bounds__(256) void ln_kernel(const float* __restrict__ x,
                                                 const float* __restrict__ g,
                                                 const float* __restrict__ bta,
                                                 u16t* __restrict__ out) {
    const int r = blockIdx.x;
    size_t src;
    if (WINDOWED) {
        const int w  = r / NWIN, n = r - w * NWIN;
        const int b_ = w >> 6, wi = w & 63;
        const int hs = (wi >> 3) * 7 + n / 7;
        const int ws = (wi & 7) * 7 + n % 7;
        int sh = hs + 3; if (sh >= 56) sh -= 56;
        int sw = ws + 3; if (sw >= 56) sw -= 56;
        src = ((size_t)b_ * 3136 + sh * 56 + sw) * DIMC;
    } else {
        src = (size_t)r * DIMC;
    }
    const int t = threadIdx.x;
    const float v = x[src + t];
    float s = v, s2 = v * v;
#pragma unroll
    for (int off = 16; off; off >>= 1) {
        s  += __shfl_xor(s, off);
        s2 += __shfl_xor(s2, off);
    }
    __shared__ float rs[8], rs2[8];
    if ((t & 31) == 0) { rs[t >> 5] = s; rs2[t >> 5] = s2; }
    __syncthreads();
    float S = 0.f, S2 = 0.f;
#pragma unroll
    for (int i = 0; i < 8; ++i) { S += rs[i]; S2 += rs2[i]; }
    const float mu  = S * (1.f / 256.f);
    const float var = S2 * (1.f / 256.f) - mu * mu;
    const float o   = (v - mu) * rsqrtf(var + LNEPS) * g[t] + bta[t];
    out[(size_t)r * DIMC + t] = f2bf(o);
}

// -------------------------------------------------------------------------
// bf16 WMMA GEMM: C[M,NOUT] = A[M,KDIM] * BT[NOUT,KDIM]^T (+ fused epilogue)
// Block: 256 thr = 8 waves; block tile 128x128; wave tile 32x64 (2x4 accums).
// KDIM/NOUT compile-time -> K loop unrolled by 8 (full for KDIM=256) or by 4 (KDIM=1024), load hoisting.
// MODE 0: qkv   -> +bias, q*=scale, store bf16
// MODE 1: proj  -> +bias, window-reverse + unshift, += x, store f32
// MODE 2: fc1   -> +bias, GELU, store bf16
// MODE 3: fc2   -> +bias, += x_res, store f32 (d_out)
// -------------------------------------------------------------------------
template <int MODE, int KDIM, int NOUT>
__global__ __launch_bounds__(256) void gemm_bf16(const u16t* __restrict__ A,
                                                 const u16t* __restrict__ BT,
                                                 const float* __restrict__ bias,
                                                 u16t* __restrict__ outb,
                                                 float* __restrict__ outf,
                                                 const float* __restrict__ resid) {
    const int lane = threadIdx.x & 31;
    const int wave = threadIdx.x >> 5;
    constexpr int NBN = NOUT >> 7;
    const int bm   = blockIdx.x / NBN;
    const int bn   = blockIdx.x - bm * NBN;
    const int tM   = bm * 128 + (wave & 3) * 32;
    const int tN   = bn * 128 + (wave >> 2) * 64;
    const int ln   = lane & 15;
    const int kl   = (lane >> 4) << 3;   // K-chunk base: 0 or 8

    const u16t* arow0 = A + (size_t)(tM + ln) * KDIM + kl;
    const u16t* arow1 = A + (size_t)(tM + 16 + ln) * KDIM + kl;
    const u16t* brow0 = BT + (size_t)(tN + ln) * KDIM + kl;
    const u16t* brow1 = BT + (size_t)(tN + 16 + ln) * KDIM + kl;
    const u16t* brow2 = BT + (size_t)(tN + 32 + ln) * KDIM + kl;
    const u16t* brow3 = BT + (size_t)(tN + 48 + ln) * KDIM + kl;

    v8f acc[2][4];
#pragma unroll
    for (int i = 0; i < 2; ++i)
#pragma unroll
        for (int j = 0; j < 4; ++j) acc[i][j] = zero_acc();

#pragma unroll (KDIM == 256 ? 8 : 4)
    for (int kk = 0; kk < KDIM; kk += 32) {
        Frag a0 = load_frag(arow0 + kk);
        Frag a1 = load_frag(arow1 + kk);
        Frag b0 = load_frag(brow0 + kk);
        Frag b1 = load_frag(brow1 + kk);
        Frag b2 = load_frag(brow2 + kk);
        Frag b3 = load_frag(brow3 + kk);
        acc[0][0] = __builtin_amdgcn_wmma_f32_16x16x32_bf16(false, a0.v, false, b0.v, (short)0, acc[0][0], false, false);
        acc[0][1] = __builtin_amdgcn_wmma_f32_16x16x32_bf16(false, a0.v, false, b1.v, (short)0, acc[0][1], false, false);
        acc[0][2] = __builtin_amdgcn_wmma_f32_16x16x32_bf16(false, a0.v, false, b2.v, (short)0, acc[0][2], false, false);
        acc[0][3] = __builtin_amdgcn_wmma_f32_16x16x32_bf16(false, a0.v, false, b3.v, (short)0, acc[0][3], false, false);
        acc[1][0] = __builtin_amdgcn_wmma_f32_16x16x32_bf16(false, a1.v, false, b0.v, (short)0, acc[1][0], false, false);
        acc[1][1] = __builtin_amdgcn_wmma_f32_16x16x32_bf16(false, a1.v, false, b1.v, (short)0, acc[1][1], false, false);
        acc[1][2] = __builtin_amdgcn_wmma_f32_16x16x32_bf16(false, a1.v, false, b2.v, (short)0, acc[1][2], false, false);
        acc[1][3] = __builtin_amdgcn_wmma_f32_16x16x32_bf16(false, a1.v, false, b3.v, (short)0, acc[1][3], false, false);
    }

#pragma unroll
    for (int i = 0; i < 2; ++i) {
#pragma unroll
        for (int j = 0; j < 4; ++j) {
            const int col = tN + 16 * j + ln;
            const float bv = bias[col];
#pragma unroll
            for (int e = 0; e < 8; ++e) {
                const int row = tM + 16 * i + ((lane >> 4) << 3) + e;
                float val = acc[i][j][e] + bv;
                if (MODE == 0) {
                    if (col < DIMC) val *= QSCALE;
                    outb[(size_t)row * NOUT + col] = f2bf(val);
                } else if (MODE == 1) {
                    const int w  = row / NWIN, n = row - w * NWIN;
                    const int b_ = w >> 6, wi = w & 63;
                    const int hs = (wi >> 3) * 7 + n / 7;
                    const int ws = (wi & 7) * 7 + n % 7;
                    int oh = hs + 3; if (oh >= 56) oh -= 56;
                    int ow = ws + 3; if (ow >= 56) ow -= 56;
                    const size_t o = ((size_t)b_ * 3136 + oh * 56 + ow) * DIMC + col;
                    outf[o] = resid[o] + val;
                } else if (MODE == 2) {
                    outb[(size_t)row * NOUT + col] = f2bf(gelu_f(val));
                } else {
                    const size_t o = (size_t)row * DIMC + col;
                    outf[o] = resid[o] + val;
                }
            }
        }
    }
}

// -------------------------------------------------------------------------
// Windowed attention: one block per (window, head); 128 thr = 4 waves.
// S = q k^T (1 WMMA/tile, K=HD=32), inline rel-pos bias + shift mask,
// shfl softmax, P via LDS -> A-frags, O = P V via LDS-staged V^T.
// -------------------------------------------------------------------------
__global__ __launch_bounds__(128) void attn_kernel(const u16t* __restrict__ qkv,
                                                   const float* __restrict__ rpb,
                                                   u16t* __restrict__ ctx) {
    __shared__ __attribute__((aligned(16))) u16t vT[32][64];     // V^T [d][m]
    __shared__ __attribute__((aligned(16))) u16t Pm[4][16][64];  // per-wave P
    const int pair = blockIdx.x;
    const int w = pair >> 3, h = pair & 7;
    const int tid = threadIdx.x, lane = tid & 31, wave = tid >> 5;
    const int ln = lane & 15;
    const int kl = (lane >> 4) << 3;
    const size_t rowbase = (size_t)w * NWIN * 768;

    // stage V^T into LDS (pad m in [49,64) with zeros)
    for (int idx = tid; idx < 2048; idx += 128) {
        int d = idx >> 6, m = idx & 63;
        vT[d][m] = (m < NWIN) ? qkv[rowbase + (size_t)m * 768 + 512 + h * 32 + d] : (u16t)0;
    }
    __syncthreads();

    // Q fragment for this wave's 16 rows (already pre-scaled in qkv epilogue)
    const int nq = wave * 16 + ln;
    Frag aq = (nq < NWIN) ? load_frag(qkv + rowbase + (size_t)nq * 768 + h * 32 + kl)
                          : zero_frag();

    // S = q k^T over 4 column tiles
    v8f s[4];
#pragma unroll
    for (int j = 0; j < 4; ++j) {
        const int nk = j * 16 + ln;
        Frag bk = (nk < NWIN) ? load_frag(qkv + rowbase + (size_t)nk * 768 + 256 + h * 32 + kl)
                              : zero_frag();
        s[j] = __builtin_amdgcn_wmma_f32_16x16x32_bf16(false, aq.v, false, bk.v, (short)0, zero_acc(), false, false);
    }

    // bias + mask (computed analytically), then row-wise softmax
    const int wi = w & 63, wr = wi >> 3, wc = wi & 7;
    float vals[4][8];
#pragma unroll
    for (int j = 0; j < 4; ++j) {
        const int m = j * 16 + ln;
#pragma unroll
        for (int e = 0; e < 8; ++e) {
            const int n = wave * 16 + ((lane >> 4) << 3) + e;
            float v = s[j][e];
            if (m >= NWIN) {
                v = -1e30f;
            } else if (n < NWIN) {
                const int r1 = n / 7, c1 = n - r1 * 7;
                const int r2 = m / 7, c2 = m - r2 * 7;
                v += rpb[((r1 - r2 + 6) * 13 + (c1 - c2 + 6)) * NHEAD + h];
                const int hn = wr * 7 + r1, wn = wc * 7 + c1;
                const int hm = wr * 7 + r2, wm = wc * 7 + c2;
                const int idn = (hn < 49 ? 0 : (hn < 53 ? 1 : 2)) * 3 + (wn < 49 ? 0 : (wn < 53 ? 1 : 2));
                const int idm = (hm < 49 ? 0 : (hm < 53 ? 1 : 2)) * 3 + (wm < 49 ? 0 : (wm < 53 ? 1 : 2));
                if (idn != idm) v -= 100.f;
            }
            vals[j][e] = v;
        }
    }
#pragma unroll
    for (int e = 0; e < 8; ++e) {
        float mx = fmaxf(fmaxf(vals[0][e], vals[1][e]), fmaxf(vals[2][e], vals[3][e]));
#pragma unroll
        for (int off = 1; off < 16; off <<= 1) mx = fmaxf(mx, __shfl_xor(mx, off));
        float sum = 0.f;
#pragma unroll
        for (int j = 0; j < 4; ++j) {
            vals[j][e] = __expf(vals[j][e] - mx);
            sum += vals[j][e];
        }
#pragma unroll
        for (int off = 1; off < 16; off <<= 1) sum += __shfl_xor(sum, off);
        const float inv = 1.f / sum;
        const int nl = ((lane >> 4) << 3) + e;
#pragma unroll
        for (int j = 0; j < 4; ++j) Pm[wave][nl][j * 16 + ln] = f2bf(vals[j][e] * inv);
    }
    __syncthreads();

    // O = P V : A-frags from P (LDS), B-frags from V^T (LDS), K=64 (2 steps)
    Frag ap0 = load_frag(&Pm[wave][ln][kl]);
    Frag ap1 = load_frag(&Pm[wave][ln][32 + kl]);
    v8f o0 = zero_acc(), o1 = zero_acc();
    {
        Frag bv00 = load_frag(&vT[ln][kl]);
        Frag bv01 = load_frag(&vT[16 + ln][kl]);
        Frag bv10 = load_frag(&vT[ln][32 + kl]);
        Frag bv11 = load_frag(&vT[16 + ln][32 + kl]);
        o0 = __builtin_amdgcn_wmma_f32_16x16x32_bf16(false, ap0.v, false, bv00.v, (short)0, o0, false, false);
        o0 = __builtin_amdgcn_wmma_f32_16x16x32_bf16(false, ap1.v, false, bv10.v, (short)0, o0, false, false);
        o1 = __builtin_amdgcn_wmma_f32_16x16x32_bf16(false, ap0.v, false, bv01.v, (short)0, o1, false, false);
        o1 = __builtin_amdgcn_wmma_f32_16x16x32_bf16(false, ap1.v, false, bv11.v, (short)0, o1, false, false);
    }
#pragma unroll
    for (int e = 0; e < 8; ++e) {
        const int n = wave * 16 + ((lane >> 4) << 3) + e;
        if (n < NWIN) {
            const size_t obase = ((size_t)w * NWIN + n) * DIMC + h * 32;
            ctx[obase + ln]      = f2bf(o0[e]);
            ctx[obase + 16 + ln] = f2bf(o1[e]);
        }
    }
}

// -------------------------------------------------------------------------
extern "C" void kernel_launch(void* const* d_in, const int* in_sizes, int n_in,
                              void* d_out, int out_size, void* d_ws, size_t ws_size,
                              hipStream_t stream) {
    (void)in_sizes; (void)n_in; (void)out_size; (void)ws_size;
    const float* x      = (const float*)d_in[0];
    const float* n1g    = (const float*)d_in[1];
    const float* n1b    = (const float*)d_in[2];
    const float* qkv_w  = (const float*)d_in[3];
    const float* qkv_b  = (const float*)d_in[4];
    const float* rpb    = (const float*)d_in[5];
    const float* proj_w = (const float*)d_in[6];
    const float* proj_b = (const float*)d_in[7];
    const float* n2g    = (const float*)d_in[8];
    const float* n2b    = (const float*)d_in[9];
    const float* fc1_w  = (const float*)d_in[10];
    const float* fc1_b  = (const float*)d_in[11];
    const float* fc2_w  = (const float*)d_in[12];
    const float* fc2_b  = (const float*)d_in[13];
    float* out = (float*)d_out;

    char* ws = (char*)d_ws;
    size_t off = 0;
    auto alloc = [&](size_t bytes) -> char* {
        char* p = ws + off;
        off += (bytes + 255) & ~(size_t)255;
        return p;
    };
    u16t* qkv_wT  = (u16t*)alloc((size_t)768 * 256 * 2);
    u16t* proj_wT = (u16t*)alloc((size_t)256 * 256 * 2);
    u16t* fc1_wT  = (u16t*)alloc((size_t)1024 * 256 * 2);
    u16t* fc2_wT  = (u16t*)alloc((size_t)256 * 1024 * 2);
    u16t* win     = (u16t*)alloc((size_t)ROWS * 256 * 2);
    u16t* qkvb    = (u16t*)alloc((size_t)ROWS * 768 * 2);
    u16t* ctx     = (u16t*)alloc((size_t)ROWS * 256 * 2);
    float* xres   = (float*)alloc((size_t)ROWS * 256 * 4);
    u16t* ybuf    = (u16t*)alloc((size_t)ROWS * 256 * 2);
    u16t* h1      = (u16t*)alloc((size_t)ROWS * 1024 * 2);

    transpose_w<<<(768 * 256 + 255) / 256, 256, 0, stream>>>(qkv_w, qkv_wT, 256, 768);
    transpose_w<<<(256 * 256 + 255) / 256, 256, 0, stream>>>(proj_w, proj_wT, 256, 256);
    transpose_w<<<(1024 * 256 + 255) / 256, 256, 0, stream>>>(fc1_w, fc1_wT, 256, 1024);
    transpose_w<<<(256 * 1024 + 255) / 256, 256, 0, stream>>>(fc2_w, fc2_wT, 1024, 256);

    ln_kernel<1><<<ROWS, 256, 0, stream>>>(x, n1g, n1b, win);
    gemm_bf16<0, 256, 768><<<784 * 6, 256, 0, stream>>>(win, qkv_wT, qkv_b, qkvb, nullptr, nullptr);
    attn_kernel<<<16384, 128, 0, stream>>>(qkvb, rpb, ctx);
    gemm_bf16<1, 256, 256><<<784 * 2, 256, 0, stream>>>(ctx, proj_wT, proj_b, nullptr, xres, x);
    ln_kernel<0><<<ROWS, 256, 0, stream>>>(xres, n2g, n2b, ybuf);
    gemm_bf16<2, 256, 1024><<<784 * 8, 256, 0, stream>>>(ybuf, fc1_wT, fc1_b, h1, nullptr, nullptr);
    gemm_bf16<3, 1024, 256><<<784 * 2, 256, 0, stream>>>(h1, fc2_wT, fc2_b, nullptr, out, xres);
}
